// RestoreBlock_73796128080523
// MI455X (gfx1250) — compile-verified, matrix operand order checked by reading, not run on hardware
//
#include <hip/hip_runtime.h>
#include <hip/hip_bf16.h>
#include <math.h>

typedef _Float16 f16;
typedef __attribute__((ext_vector_type(16))) _Float16 v16h;
typedef __attribute__((ext_vector_type(8)))  _Float16 h8;
typedef __attribute__((ext_vector_type(8)))  float    v8f;

#define WMMA_F16(a,b,c) __builtin_amdgcn_wmma_f32_16x16x32_f16(false,(a),false,(b),(short)0,(c),false,false)

namespace cfg {
constexpr int B=4, H=60, W=60, C=256, HEADS=8, HD=32, WS=7, WSS=3, KWS=27;
constexpr int NWD=9;                 // windows per spatial dim (63/7)
constexpr int NWIN=B*NWD*NWD;        // 324
constexpr int QP=64;                 // queries per window padded 49->64
constexpr int NS=27;                 // y spatial side
constexpr int NKEY=KWS*KWS;          // 729
constexpr int NCH=23;                // key chunks of 32 (736)
constexpr int NTOK=B*H*W;            // 14400
constexpr int MQ=NWIN*QP;            // 20736
constexpr int MY=2944;               // y rows padded 2916->2944 (mult of 32)
constexpr float NEG=-1000.f;
}

__device__ inline v8f fzero(){ v8f z;
#pragma unroll
  for(int i=0;i<8;++i) z[i]=0.f; return z; }
__device__ inline v16h hzero(){ v16h z;
#pragma unroll
  for(int i=0;i<16;++i) z[i]=(f16)0.f; return z; }

// Load one 16x32 f16 fragment (A-layout; B uses the transposed operand so the
// same contiguous pattern applies). Lane L holds row (caller-resolved) and
// k-columns  base..base+7, base+16..base+23  with base = (L>=16)?8:0.
__device__ inline v16h frag_ld(const f16* base, int row, int stride, int k0, int lane){
  int off = k0 + ((lane >> 4) << 3);
  const f16* p = base + (size_t)row * stride + off;
  union { h8 a[2]; v16h v; } u;
  u.a[0] = *(const h8*)(p);
  u.a[1] = *(const h8*)(p + 16);
  return u.v;
}

// Async global->LDS copy of 16 bytes (tracked by ASYNCcnt).
__device__ inline void async_ld16(const f16* gsrc, f16* lds){
  unsigned la = (unsigned)(uintptr_t)lds;
  asm volatile("global_load_async_to_lds_b128 %0, %1, off"
               :: "v"(la), "v"(gsrc) : "memory");
}

// Block (256 threads) layernorm over 256 channels; returns (x-m)*rsqrt(v+eps).
__device__ inline float blk_ln(float v, float* sb){
  int tid = threadIdx.x;
  sb[tid]=v; __syncthreads();
  for(int s=128;s>0;s>>=1){ if(tid<s) sb[tid]+=sb[tid+s]; __syncthreads(); }
  float m = sb[0]*(1.f/256.f); __syncthreads();
  float d = v-m;
  sb[tid]=d*d; __syncthreads();
  for(int s=128;s>0;s>>=1){ if(tid<s) sb[tid]+=sb[tid+s]; __syncthreads(); }
  float var = sb[0]*(1.f/256.f);
  return d*rsqrtf(var+1e-5f);
}

// ---------------- weight transpose + f32->f16 convert: dst[n][k]=src[k][n] --
__global__ __launch_bounds__(256) void k_tcvt(const float* __restrict__ s,
                                              f16* __restrict__ d, int K, int N){
  size_t idx = (size_t)blockIdx.x*256 + threadIdx.x;
  if(idx >= (size_t)K*N) return;
  int n = (int)(idx / K), k = (int)(idx % K);
  d[idx] = (f16)s[(size_t)k*N + n];
}

// ---------------- LN(y) -> f16, zero pad rows ------------------------------
__global__ __launch_bounds__(256) void k_ln_y(const float* __restrict__ y,
   const float* __restrict__ g, const float* __restrict__ b, f16* __restrict__ yn){
  __shared__ float sb[256];
  int t = blockIdx.x, c = threadIdx.x;
  if(t >= cfg::B*cfg::NS*cfg::NS){ yn[(size_t)t*256+c]=(f16)0.f; return; }
  float v = y[(size_t)t*256+c];
  float nv = blk_ln(v, sb);
  yn[(size_t)t*256+c] = (f16)(nv*g[c]+b[c]);
}

// ---------------- window-gathered LN(x) -> f16, zero pad rows --------------
__global__ __launch_bounds__(256) void k_ln_xwin(const float* __restrict__ x,
   const float* __restrict__ g, const float* __restrict__ b, f16* __restrict__ xn){
  using namespace cfg;
  __shared__ float sb[256];
  int row = blockIdx.x, c = threadIdx.x;
  int w = row>>6, p = row&63;
  int bb = w/(NWD*NWD), wr = w%(NWD*NWD), wh=wr/NWD, ww=wr%NWD;
  int i = p/WS, j = p%WS;
  int hhp = wh*WS+i, wwp = ww*WS+j;
  bool valid = (p < WS*WS) && hhp<H && wwp<W;
  if(!valid){ xn[(size_t)row*256+c]=(f16)0.f; return; }
  float v = x[((size_t)bb*H*W + (size_t)hhp*W + wwp)*256 + c];
  float nv = blk_ln(v, sb);
  xn[(size_t)row*256+c] = (f16)(nv*g[c]+b[c]);
}

// ---------------- generic WMMA GEMM:  C[MxN] = A[MxK] * Bt[NxK]^T ----------
// Each wave computes a 32x64 tile (2 M-tiles x 4 N-tiles): B fragments are
// reused across both M-tiles -> 24 bytes loaded per WMMA instead of 40.
// Requires M % 32 == 0, N % 64 == 0, K % 32 == 0.
__global__ __launch_bounds__(256) void k_gemm(const f16* __restrict__ A,
   const f16* __restrict__ Bt, const float* __restrict__ bias,
   f16* __restrict__ Cf16, float* __restrict__ Cf32,
   int M, int N, int K, int dogelu){
  const int lane = threadIdx.x&31, wid = threadIdx.x>>5;
  const int nt64 = N>>6;
  const int total = (M>>5)*nt64;
  int gt = blockIdx.x*8 + wid;
  if(gt>=total) return;
  const int tm = gt/nt64, tn = gt - tm*nt64;
  const int m0 = tm<<5, n0 = tn<<6;
  const int col = lane&15, hf = lane>>4;
  v8f acc[2][4];
#pragma unroll
  for(int i=0;i<2;++i)
#pragma unroll
    for(int j=0;j<4;++j) acc[i][j]=fzero();
  const int ar0 = m0 + col, ar1 = m0 + 16 + col;
  for(int k0=0;k0<K;k0+=32){
    if(k0+32<K){
      __builtin_prefetch(A + (size_t)ar0*K + k0 + 32, 0, 0);
      __builtin_prefetch(A + (size_t)ar1*K + k0 + 32, 0, 0);
    }
    v16h a0 = frag_ld(A, ar0, K, k0, lane);
    v16h a1 = frag_ld(A, ar1, K, k0, lane);
#pragma unroll
    for(int j=0;j<4;++j){
      v16h b = frag_ld(Bt, n0 + j*16 + col, K, k0, lane);
      acc[0][j] = WMMA_F16(a0,b,acc[0][j]);
      acc[1][j] = WMMA_F16(a1,b,acc[1][j]);
    }
  }
#pragma unroll
  for(int i=0;i<2;++i)
#pragma unroll
  for(int j=0;j<4;++j){
    int n = n0 + j*16 + col;
    float bv = bias ? bias[n] : 0.f;
#pragma unroll
    for(int r=0;r<8;++r){
      int m = m0 + i*16 + r + 8*hf;
      float v = acc[i][j][r] + bv;
      if(dogelu) v = 0.5f*v*(1.f+erff(v*0.70710678118654752f));
      size_t idx = (size_t)m*N + n;
      if(Cf32) Cf32[idx] = v;
      if(Cf16) Cf16[idx] = (f16)v;
    }
  }
}

// ---------------- fused window cross-attention (flash-style) ---------------
// 1 block per window; 8 waves = 8 heads. Keys in chunks of 32. V rows for the
// NEXT chunk are prefetched global->LDS with async-LDS loads (ASYNCcnt) into a
// double-buffered row-major staging area, then register-transposed into the
// B-fragment-friendly Vt layout while WMMAs of the current chunk execute.
__global__ __launch_bounds__(256) void k_attn(const f16* __restrict__ Q,
                                              const f16* __restrict__ KV,
                                              f16* __restrict__ O){
  using namespace cfg;
  __shared__ __align__(16) f16 Vrow[2][32][256];  // async staging (2x16KB)
  __shared__ __align__(16) f16 Vt[C][32];         // V chunk, [channel][keyslot]
  __shared__ __align__(16) f16 Pb[8][16][32];     // per-wave P tile
  const int tid=threadIdx.x, lane=tid&31, hd=tid>>5;
  const int w = blockIdx.x;
  const int b = w/(NWD*NWD), wr = w%(NWD*NWD), wh = wr/NWD, ww = wr%NWD;
  const int col = lane&15, hf = lane>>4;
  const float scale = 0.17677669529663687f;   // 1/sqrt(32)

  // this thread's staging slot: one V row (token) x 32 channels
  const int slot = tid>>3, cg = (tid&7)*32;

  v16h qf[4];
#pragma unroll
  for(int qt=0;qt<4;++qt)
    qf[qt] = frag_ld(Q, w*QP + qt*16 + col, C, hd*HD, lane);

  float rm[4][8], rl[4][8];
  v8f oacc[4][2];
#pragma unroll
  for(int qt=0;qt<4;++qt){
#pragma unroll
    for(int r=0;r<8;++r){ rm[qt][r]=-1e30f; rl[qt][r]=0.f; }
    oacc[qt][0]=fzero(); oacc[qt][1]=fzero();
  }

  // issue async staging for chunk 0
  {
    int kk = slot;                       // c0 = 0
    int ii = kk/KWS, jj = kk - ii*KWS;
    int hc = wh*WSS + ii - 12, wc = ww*WSS + jj - 12;
    bool valid = (kk<NKEY) && hc>=0 && hc<NS && wc>=0 && wc<NS;
    if(valid){
      const f16* src = KV + (size_t)((b*NS+hc)*NS+wc)*512 + 256 + cg;
      f16* dst = &Vrow[0][slot][cg];
#pragma unroll
      for(int c4=0;c4<32;c4+=8) async_ld16(src+c4, dst+c4);
    }
  }

  for(int chnk=0; chnk<NCH; ++chnk){
    const int c0 = chnk*32;
    asm volatile("s_wait_asynccnt 0" ::: "memory");  // my staged rows arrived
    __syncthreads();                    // prev chunk fully consumed (Vt free)
    { // register-transpose my staged row into Vt
      int kk = c0 + slot;
      int ii = kk/KWS, jj = kk - ii*KWS;
      int hc = wh*WSS + ii - 12, wc = ww*WSS + jj - 12;
      bool valid = (kk<NKEY) && hc>=0 && hc<NS && wc>=0 && wc<NS;
      const f16* src = &Vrow[chnk&1][slot][cg];
#pragma unroll
      for(int c4=0;c4<32;c4+=8){
        h8 vv;
        if(valid) vv = *(const h8*)(src + c4);
        else {
#pragma unroll
          for(int e=0;e<8;++e) vv[e]=(f16)0.f;
        }
#pragma unroll
        for(int e=0;e<8;++e) Vt[cg+c4+e][slot] = vv[e];
      }
    }
    if(chnk+1 < NCH){ // issue async staging for next chunk into other buffer
      int kk = c0 + 32 + slot;
      int ii = kk/KWS, jj = kk - ii*KWS;
      int hc = wh*WSS + ii - 12, wc = ww*WSS + jj - 12;
      bool valid = (kk<NKEY) && hc>=0 && hc<NS && wc>=0 && wc<NS;
      if(valid){
        const f16* src = KV + (size_t)((b*NS+hc)*NS+wc)*512 + 256 + cg;
        f16* dst = &Vrow[(chnk+1)&1][slot][cg];
#pragma unroll
        for(int c4=0;c4<32;c4+=8) async_ld16(src+c4, dst+c4);
      }
    }
    __syncthreads();                    // Vt published to all waves

    v16h kf[2]; float mv[2];
#pragma unroll
    for(int t=0;t<2;++t){
      int kk = c0 + t*16 + col;
      int ii = kk/KWS, jj = kk - ii*KWS;
      int hc = wh*WSS + ii - 12, wc = ww*WSS + jj - 12;
      bool valid = (kk<NKEY) && hc>=0 && hc<NS && wc>=0 && wc<NS;
      bool corner = valid && (hc>=NS-3) && (wc>=NS-3);
      mv[t] = (!valid || corner) ? NEG : 0.f;
      if(valid) kf[t] = frag_ld(KV, (b*NS+hc)*NS+wc, 512, hd*HD, lane);
      else      kf[t] = hzero();
    }

#pragma unroll
    for(int qt=0;qt<4;++qt){
      v8f s0 = WMMA_F16(qf[qt], kf[0], fzero());
      v8f s1 = WMMA_F16(qf[qt], kf[1], fzero());
      float p0[8], p1[8];
#pragma unroll
      for(int r=0;r<8;++r){
        float a  = s0[r]*scale + mv[0];
        float bb = s1[r]*scale + mv[1];
        float mx = fmaxf(a,bb);
#pragma unroll
        for(int d=8; d>=1; d>>=1) mx = fmaxf(mx, __shfl_xor(mx, d, 16));
        float nm = fmaxf(rm[qt][r], mx);
        float corr = __expf(rm[qt][r]-nm);
        float e0 = __expf(a-nm), e1 = __expf(bb-nm);
        float sm = e0+e1;
#pragma unroll
        for(int d=8; d>=1; d>>=1) sm += __shfl_xor(sm, d, 16);
        rl[qt][r] = rl[qt][r]*corr + sm;
        rm[qt][r] = nm;
        p0[r]=e0; p1[r]=e1;
        oacc[qt][0][r] *= corr;
        oacc[qt][1][r] *= corr;
      }
#pragma unroll
      for(int r=0;r<8;++r){
        Pb[hd][r+8*hf][col]    = (f16)p0[r];
        Pb[hd][r+8*hf][col+16] = (f16)p1[r];
      }
      asm volatile("s_wait_dscnt 0" ::: "memory");   // P store -> A-frag read
      v16h pf = frag_ld(&Pb[hd][0][0], col, 32, 0, lane);
#pragma unroll
      for(int nt=0;nt<2;++nt){
        v16h vf = frag_ld(&Vt[0][0], hd*HD + nt*16 + col, 32, 0, lane);
        oacc[qt][nt] = WMMA_F16(pf, vf, oacc[qt][nt]);
      }
    }
  }
#pragma unroll
  for(int qt=0;qt<4;++qt)
#pragma unroll
    for(int nt=0;nt<2;++nt)
#pragma unroll
      for(int r=0;r<8;++r){
        float v = oacc[qt][nt][r] / rl[qt][r];
        O[(size_t)(w*QP + qt*16 + r + 8*hf)*C + hd*HD + nt*16 + col] = (f16)v;
      }
}

// ---------------- xo = x + ls1*(proj) ; LN2 -> f16 -------------------------
__global__ __launch_bounds__(256) void k_xo_ln2(const float* __restrict__ x,
   const float* __restrict__ pj, const float* __restrict__ ls1,
   const float* __restrict__ g2, const float* __restrict__ b2,
   float* __restrict__ xo, f16* __restrict__ xn2){
  using namespace cfg;
  __shared__ float sb[256];
  int tkn = blockIdx.x, c = threadIdx.x;
  int bb = tkn/(H*W), r = tkn%(H*W);
  int hhp = r/W, wwp = r%W;
  int wh=hhp/WS, i=hhp%WS, wn=wwp/WS, j=wwp%WS;
  int w = (bb*NWD + wh)*NWD + wn;
  size_t prow = (size_t)w*QP + i*WS + j;
  float pv = pj[prow*256 + c];            // proj output (bias already added)
  float xv = x[(size_t)tkn*256 + c];      // shortcut = original x
  float xov = xv + ls1[c]*pv;
  xo[(size_t)tkn*256+c] = xov;
  float nv = blk_ln(xov, sb);
  xn2[(size_t)tkn*256+c] = (f16)(nv*g2[c]+b2[c]);
}

// ---------------- final residual: out = xo + ls2*fc2out --------------------
__global__ __launch_bounds__(256) void k_out(const float* __restrict__ xo,
   const float* __restrict__ f2o, const float* __restrict__ ls2,
   float* __restrict__ out){
  size_t idx = (size_t)blockIdx.x*256 + threadIdx.x;
  int c = threadIdx.x;
  out[idx] = xo[idx] + ls2[c]*f2o[idx];
}

extern "C" void kernel_launch(void* const* d_in, const int* in_sizes, int n_in,
                              void* d_out, int out_size, void* d_ws, size_t ws_size,
                              hipStream_t stream){
  using namespace cfg;
  const float* x  =(const float*)d_in[0];
  const float* y  =(const float*)d_in[1];
  const float* n1g=(const float*)d_in[2];
  const float* n1b=(const float*)d_in[3];
  const float* qw =(const float*)d_in[4];
  const float* kvw=(const float*)d_in[5];
  const float* pw =(const float*)d_in[6];
  const float* pb =(const float*)d_in[7];
  const float* n2g=(const float*)d_in[8];
  const float* n2b=(const float*)d_in[9];
  const float* f1w=(const float*)d_in[10];
  const float* f1b=(const float*)d_in[11];
  const float* f2w=(const float*)d_in[12];
  const float* f2b=(const float*)d_in[13];
  const float* ls1=(const float*)d_in[14];
  const float* ls2=(const float*)d_in[15];
  float* out=(float*)d_out;
  char* ws=(char*)d_ws;
  const size_t MB = 1024*1024;

  // transposed f16 weights
  f16* qT  = (f16*)(ws + 0);
  f16* kvT = qT  + 256*256;
  f16* pT  = kvT + 512*256;
  f16* f1T = pT  + 256*256;
  f16* f2T = f1T + 1024*256;
  // persistent activations
  f16*   kvF = (f16*)(ws + 2*MB);    // [2944][512]
  f16*   oF  = (f16*)(ws + 6*MB);    // [20736][256]
  float* xo  = (float*)(ws + 17*MB); // [14400][256]
  f16*   xn2 = (f16*)(ws + 32*MB);   // [14400][256]
  // reused region @40MB (sequential lifetimes, stream-ordered)
  f16*   yn  = (f16*)(ws + 40*MB);   // [2944][256]
  f16*   xnw = (f16*)(ws + 42*MB);   // [20736][256]
  f16*   qF  = (f16*)(ws + 53*MB);   // [20736][256]
  float* pj  = (float*)(ws + 40*MB); // [20736][256] (after attn)
  f16*   h1  = (f16*)(ws + 40*MB);   // [14400][1024] (after xo_ln2)
  float* f2o = (float*)(ws + 70*MB); // [14400][256]

  // weights -> f16 transposed
  k_tcvt<<<256 ,256,0,stream>>>(qw , qT , 256 , 256 );
  k_tcvt<<<512 ,256,0,stream>>>(kvw, kvT, 256 , 512 );
  k_tcvt<<<256 ,256,0,stream>>>(pw , pT , 256 , 256 );
  k_tcvt<<<1024,256,0,stream>>>(f1w, f1T, 256 , 1024);
  k_tcvt<<<1024,256,0,stream>>>(f2w, f2T, 1024, 256 );
  // layernorm1 (y once per token — kills the 62 GFLOP per-window recompute)
  k_ln_y   <<<MY,256,0,stream>>>(y, n1g, n1b, yn);
  k_ln_xwin<<<MQ,256,0,stream>>>(x, n1g, n1b, xnw);
  // projections
  k_gemm<<<(MQ/32*(256/64)+7)/8,256,0,stream>>>(xnw, qT , nullptr, qF , nullptr, MQ ,256,256,0);
  k_gemm<<<(MY/32*(512/64)+7)/8,256,0,stream>>>(yn , kvT, nullptr, kvF, nullptr, MY ,512,256,0);
  // fused attention
  k_attn<<<NWIN,256,0,stream>>>(qF, kvF, oF);
  // output projection (+bias)
  k_gemm<<<(MQ/32*(256/64)+7)/8,256,0,stream>>>(oF, pT, pb, nullptr, pj, MQ,256,256,0);
  // residual + LN2
  k_xo_ln2<<<NTOK,256,0,stream>>>(x, pj, ls1, n2g, n2b, xo, xn2);
  // MLP
  k_gemm<<<(NTOK/32*(1024/64)+7)/8,256,0,stream>>>(xn2, f1T, f1b, h1 , nullptr, NTOK,1024,256 ,1);
  k_gemm<<<(NTOK/32*(256/64)+7)/8 ,256,0,stream>>>(h1 , f2T, f2b, nullptr, f2o, NTOK,256 ,1024,0);
  // final residual
  k_out<<<NTOK,256,0,stream>>>(xo, f2o, ls2, out);
}
